// NMN_6554120093798
// MI455X (gfx1250) — matrix operand order repaired and checked by reading, not compiled, not run on hardware
//
#include <hip/hip_runtime.h>

typedef __attribute__((ext_vector_type(16))) __bf16 v16bf;
typedef __attribute__((ext_vector_type(8)))  float  v8f;

#define EPSF   1e-12f
#define SCALEF 0.03608439182435161f   // 1/sqrt(768)

// ---- workspace offsets (in floats) ----
#define O_CFIND 0
#define O_QFIND 49152
#define O_VFIND 98304
#define O_CTR   147456
#define O_QTR   196608
#define O_VTR   245760
#define O_CD1   294912
#define O_CD2   344064
#define O_EPTR  393216
#define O_EBW   442368
#define O_Z1    491520
#define O_Z2    638976
#define O_MEM1  786432
#define O_MEM2  835584
#define O_RP    884736
#define O_RPB   950272
#define O_T1F   1015808
#define O_T2F   1081344
#define O_W1    1146880
#define O_W2    1212416
#define O_SCF   1277952
#define O_SCT   1343488
#define O_SCS   1409024
#define O_PB    1474560
#define O_PF    1475072
#define O_PFB   1475584
#define O_C1F   1476096
#define O_C2F   1476160
#define O_QKBF  1476224
#define O_QKBT  1476288
#define O_EPART 1476352
// total: 1,869,568 floats (~7.5 MB)

// ---------------- WMMA bf16 GEMM, 2x2 tile blocking per wave -------------------
//  C(MxN) = A(MxK) * B(KxN) [+bias],  BT=1 means B is stored transposed (NxK).
//  Each wave computes a 32x32 block of C: 4 v_wmma per k-step, 2x operand reuse.
template <int BT>
__global__ void __launch_bounds__(128)
k_gemm_wmma(const float* __restrict__ A, const float* __restrict__ B,
            const float* __restrict__ bias, float* __restrict__ C,
            int M, int N, int K)
{
    const int wave = threadIdx.x >> 5;
    const int lane = threadIdx.x & 31;
    const int Mt2 = M >> 5, Nt2 = N >> 5;          // 32x32 blocks
    const int task = blockIdx.x * 4 + wave;
    if (task >= Mt2 * Nt2) return;                 // wave-uniform: EXEC stays full
    const int mt2 = task % Mt2, nt2 = task / Mt2;
    const int r0 = (mt2 << 5) + (lane & 15);
    const int c0 = (nt2 << 5) + (lane & 15);
    const int kh = lane >> 4;

    v8f acc00 = {}, acc01 = {}, acc10 = {}, acc11 = {};
    const float* A0 = A + (size_t)r0 * K + (kh << 4);
    const float* A1 = A0 + (size_t)16 * K;
    for (int k0 = 0; k0 < K; k0 += 32) {
        v16bf a0, a1, b0, b1;
        // A 16x32 fragment: lane row = lane&15, K(e) = k0 + 16*kh + e
        #pragma unroll
        for (int e = 0; e < 16; ++e) {
            a0[e] = (__bf16)A0[k0 + e];
            a1[e] = (__bf16)A1[k0 + e];
        }
        // B 32x16 fragment: lane col = lane&15, K(e) = k0 + 16*kh + e
        if (BT) {
            const float* B0 = B + (size_t)c0 * K + k0 + (kh << 4);
            const float* B1 = B0 + (size_t)16 * K;
            #pragma unroll
            for (int e = 0; e < 16; ++e) { b0[e] = (__bf16)B0[e]; b1[e] = (__bf16)B1[e]; }
        } else {
            const float* B0 = B + (size_t)(k0 + (kh << 4)) * N + c0;
            #pragma unroll
            for (int e = 0; e < 16; ++e) {
                b0[e] = (__bf16)B0[(size_t)e * N];
                b1[e] = (__bf16)B0[(size_t)e * N + 16];
            }
        }
        acc00 = __builtin_amdgcn_wmma_f32_16x16x32_bf16(false, a0, false, b0, (short)0, acc00, false, false);
        acc01 = __builtin_amdgcn_wmma_f32_16x16x32_bf16(false, a0, false, b1, (short)0, acc01, false, false);
        acc10 = __builtin_amdgcn_wmma_f32_16x16x32_bf16(false, a1, false, b0, (short)0, acc10, false, false);
        acc11 = __builtin_amdgcn_wmma_f32_16x16x32_bf16(false, a1, false, b1, (short)0, acc11, false, false);
    }
    const float bb0 = bias ? bias[c0]      : 0.0f;
    const float bb1 = bias ? bias[c0 + 16] : 0.0f;
    #pragma unroll
    for (int r = 0; r < 8; ++r) {
        const int m0 = (mt2 << 5) + (kh << 3) + r;   // D: VGPR r -> M=r (+8 upper lanes)
        const int m1 = m0 + 16;
        C[(size_t)m0 * N + c0]      = acc00[r] + bb0;
        C[(size_t)m0 * N + c0 + 16] = acc01[r] + bb1;
        C[(size_t)m1 * N + c0]      = acc10[r] + bb0;
        C[(size_t)m1 * N + c0 + 16] = acc11[r] + bb1;
    }
}

// ---------------- pointer transforms, ptr_avg softmax output -------------------
__global__ void k_ptr(const float* __restrict__ P_in, const float* __restrict__ mp_in,
                      float* __restrict__ ws, float* __restrict__ out)
{
    const int n = threadIdx.x;
    if (n >= 64) return;
    float P[8], pf[8], Pb[8], pfb[8];
    #pragma unroll
    for (int l = 0; l < 8; ++l) P[l] = P_in[n*8 + l];
    pf[0] = 0.f;
    #pragma unroll
    for (int l = 1; l < 8; ++l) pf[l] = P[l-1];
    pf[7] += P[7];
    #pragma unroll
    for (int l = 0; l < 7; ++l) Pb[l] = P[l+1];
    Pb[7] = 0.f; Pb[0] += P[0];
    #pragma unroll
    for (int l = 0; l < 7; ++l) pfb[l] = pf[l+1];
    pfb[7] = 0.f; pfb[0] += pf[0];
    float c1 = 0.f, c2 = 0.f;
    #pragma unroll
    for (int l = 0; l < 8; ++l) { c2 += pf[l]*pf[l]; c1 += pf[l]*pfb[l]; }
    float mp[9];
    #pragma unroll
    for (int m = 0; m < 9; ++m) mp[m] = mp_in[n*9 + m];
    // module ptrs: m0,m2,m7->P ; m1,m6->pf ; m3->pfb ; m4,m5,m8->Pb
    const float wP = mp[0]+mp[2]+mp[7], wpf = mp[1]+mp[6], wpfb = mp[3], wPb = mp[4]+mp[5]+mp[8];
    float pa[8]; float mx = -1e30f;
    #pragma unroll
    for (int l = 0; l < 8; ++l) {
        pa[l] = wP*P[l] + wpf*pf[l] + wpfb*pfb[l] + wPb*Pb[l];
        mx = fmaxf(mx, pa[l]);
    }
    float sum = 0.f;
    #pragma unroll
    for (int l = 0; l < 8; ++l) { pa[l] = expf((pa[l]-mx)*10.0f); sum += pa[l]; }
    const float inv = 1.0f/sum;
    #pragma unroll
    for (int l = 0; l < 8; ++l) {
        out[524288 + n*8 + l] = pa[l]*inv;
        ws[O_PB +n*8+l] = Pb[l];
        ws[O_PF +n*8+l] = pf[l];
        ws[O_PFB+n*8+l] = pfb[l];
    }
    ws[O_C1F+n] = c1; ws[O_C2F+n] = c2;
}

// ---------------- per-(n,s) stack reads ---------------------------------------
__global__ void __launch_bounds__(256)
k_attread(const float* __restrict__ S, const float* __restrict__ P_in,
          float* __restrict__ ws)
{
    const int tid = blockIdx.x*256 + threadIdx.x;    // 65536
    const int n = tid >> 10;
    const float* Srow = S + (size_t)tid * 8;
    float rp = 0.f, rpb = 0.f, a1 = 0.f, a2 = 0.f;
    #pragma unroll
    for (int l = 0; l < 8; ++l) {
        const float s  = Srow[l];
        const float P  = P_in[n*8+l];
        const float Pb = ws[O_PB +n*8+l];
        const float pf = ws[O_PF +n*8+l];
        const float pfb= ws[O_PFB+n*8+l];
        rp  += s*P;
        rpb += s*Pb;
        a2  += s*pf *(1.f-pf);   // read(find_stack, pf)  linear part
        a1  += s*pfb*(1.f-pf);   // read(find_stack, pfb) linear part
    }
    ws[O_RP +tid] = rp;  ws[O_RPB+tid] = rpb;
    ws[O_T2F+tid] = a2;  ws[O_T1F+tid] = a1;
}

// ---------------- spatial softmax over s (1024) for r_P and r_Pb --------------
__global__ void __launch_bounds__(256)
k_softmax(float* __restrict__ ws)
{
    __shared__ float red[256];
    const int n = blockIdx.x, t = threadIdx.x;
    for (int which = 0; which < 2; ++which) {
        const float* src = ws + (which ? O_RPB : O_RP) + n*1024;
        float*       dst = ws + (which ? O_W1  : O_W2) + n*1024;
        float4 v = reinterpret_cast<const float4*>(src)[t];
        float mx = fmaxf(fmaxf(v.x,v.y), fmaxf(v.z,v.w));
        red[t] = mx; __syncthreads();
        for (int s2 = 128; s2; s2 >>= 1) { if (t < s2) red[t] = fmaxf(red[t], red[t+s2]); __syncthreads(); }
        mx = red[0]; __syncthreads();
        float4 e = make_float4(expf(v.x-mx), expf(v.y-mx), expf(v.z-mx), expf(v.w-mx));
        red[t] = e.x+e.y+e.z+e.w; __syncthreads();
        for (int s2 = 128; s2; s2 >>= 1) { if (t < s2) red[t] += red[t+s2]; __syncthreads(); }
        const float inv = 1.0f/red[0]; __syncthreads();
        reinterpret_cast<float4*>(dst)[t] = make_float4(e.x*inv, e.y*inv, e.z*inv, e.w*inv);
    }
}

// ---------------- q . k_bias dots ---------------------------------------------
__global__ void k_qkb(float* __restrict__ ws, const float* __restrict__ fkb,
                      const float* __restrict__ tkb)
{
    const int n = threadIdx.x;
    if (n >= 64) return;
    const float* qf = ws + O_QFIND + n*768;
    const float* qt = ws + O_QTR   + n*768;
    float sf = 0.f, st = 0.f;
    for (int d = 0; d < 768; ++d) { sf += qf[d]*fkb[d]; st += qt[d]*tkb[d]; }
    ws[O_QKBF+n] = sf; ws[O_QKBT+n] = st;
}

// ---------------- single streaming pass over image_hidden (201 MB) ------------
__global__ void __launch_bounds__(256)
k_stream(const float* __restrict__ X, const float* __restrict__ mask,
         const float* __restrict__ scene_w, const float* __restrict__ scene_b,
         float* __restrict__ ws)
{
    __shared__ __align__(16) float vf_s[768];
    __shared__ __align__(16) float vt_s[768];
    __shared__ __align__(16) float sw_s[768];
    __shared__ float e_sh[1536];
    const int bx = blockIdx.x;
    const int n = bx >> 2, chunk = bx & 3;
    const int t = threadIdx.x, w = t >> 5, lane = t & 31;
    const float* vf = ws + O_VFIND + n*768;
    const float* vt = ws + O_VTR   + n*768;
    for (int i = t; i < 768; i += 256) { vf_s[i] = vf[i]; vt_s[i] = vt[i]; sw_s[i] = scene_w[i]; }
    for (int i = t; i < 1536; i += 256) e_sh[i] = 0.f;
    __syncthreads();
    const float qkbf = ws[O_QKBF+n], qkbt = ws[O_QKBT+n], sb = scene_b[0];
    float4 ep[6], eb[6];
    #pragma unroll
    for (int j = 0; j < 6; ++j) { ep[j] = make_float4(0,0,0,0); eb[j] = make_float4(0,0,0,0); }

    for (int it = 0; it < 32; ++it) {
        const int s   = chunk*256 + it*8 + w;
        const int gsi = n*1024 + s;
        const float4* xr = reinterpret_cast<const float4*>(X + (size_t)gsi * 768);
        float4 x[6];
        float nrm2 = 0.f, dvf = 0.f, dvt = 0.f, dsc = 0.f;
        #pragma unroll
        for (int j = 0; j < 6; ++j) {
            const int f4 = j*32 + lane;
            const float4 xv = xr[f4];
            x[j] = xv;
            const float4 a = reinterpret_cast<const float4*>(vf_s)[f4];
            const float4 b = reinterpret_cast<const float4*>(vt_s)[f4];
            const float4 c = reinterpret_cast<const float4*>(sw_s)[f4];
            nrm2 += xv.x*xv.x + xv.y*xv.y + xv.z*xv.z + xv.w*xv.w;
            dvf  += xv.x*a.x  + xv.y*a.y  + xv.z*a.z  + xv.w*a.w;
            dvt  += xv.x*b.x  + xv.y*b.y  + xv.z*b.z  + xv.w*b.w;
            dsc  += xv.x*c.x  + xv.y*c.y  + xv.z*c.z  + xv.w*c.w;
        }
        #pragma unroll
        for (int off = 16; off; off >>= 1) {
            nrm2 += __shfl_xor(nrm2, off);
            dvf  += __shfl_xor(dvf,  off);
            dvt  += __shfl_xor(dvt,  off);
            dsc  += __shfl_xor(dsc,  off);
        }
        const float wt1 = ws[O_W1+gsi];
        const float wt2 = ws[O_W2+gsi];
        #pragma unroll
        for (int j = 0; j < 6; ++j) {
            ep[j].x += wt2*x[j].x; ep[j].y += wt2*x[j].y; ep[j].z += wt2*x[j].z; ep[j].w += wt2*x[j].w;
            eb[j].x += wt1*x[j].x; eb[j].y += wt1*x[j].y; eb[j].z += wt1*x[j].z; eb[j].w += wt1*x[j].w;
        }
        if (lane == 0) {
            const float nrm = sqrtf(nrm2);
            const float mk  = mask[gsi];
            ws[O_SCF+gsi] = (dvf / fmaxf(nrm, EPSF) + qkbf) * SCALEF + mk;
            const float dent = fmaxf(wt2 * nrm, EPSF);
            ws[O_SCT+gsi] = ((wt2 * dvt) / dent + qkbt) * SCALEF + mk;
            ws[O_SCS+gsi] = dsc + sb;
        }
    }
    // deterministic cross-wave reduction of the extract accumulators
    for (int wv = 0; wv < 8; ++wv) {
        if (w == wv) {
            #pragma unroll
            for (int j = 0; j < 6; ++j) {
                const int d0 = (j*32 + lane)*4;
                e_sh[d0+0] += ep[j].x; e_sh[d0+1] += ep[j].y; e_sh[d0+2] += ep[j].z; e_sh[d0+3] += ep[j].w;
                e_sh[768+d0+0] += eb[j].x; e_sh[768+d0+1] += eb[j].y; e_sh[768+d0+2] += eb[j].z; e_sh[768+d0+3] += eb[j].w;
            }
        }
        __syncthreads();
    }
    float* dst = ws + O_EPART + (size_t)(n*4 + chunk)*1536;
    for (int i = t; i < 1536; i += 256) dst[i] = e_sh[i];
}

// ---------------- reduce the 4 partial extract blocks per batch ----------------
__global__ void k_ered(float* __restrict__ ws)
{
    const int tid = blockIdx.x*256 + threadIdx.x;  // 49152
    const int n = tid / 768, d = tid % 768;
    const float* base = ws + O_EPART + (size_t)n*6144;
    float sp = 0.f, sb = 0.f;
    #pragma unroll
    for (int c = 0; c < 4; ++c) { sp += base[c*1536 + d]; sb += base[c*1536 + 768 + d]; }
    ws[O_EPTR+tid] = sp;
    ws[O_EBW +tid] = sb;
}

// ---------------- build concat inputs for the memory GEMMs --------------------
__global__ void k_concat(const float* __restrict__ cs, float* __restrict__ ws)
{
    const int tid = blockIdx.x*256 + threadIdx.x;  // 49152
    const int n = tid / 768, d = tid % 768;
    const float c   = cs[tid];
    const float epv = ws[O_EPTR+tid], ebv = ws[O_EBW+tid];
    const float cd1 = ws[O_CD1+tid],  cd2 = ws[O_CD2+tid];
    const size_t z = (size_t)n*2304 + d;
    ws[O_Z1+z] = c;  ws[O_Z1+z+768] = epv;      ws[O_Z1+z+1536] = epv*cd1;
    ws[O_Z2+z] = c;  ws[O_Z2+z+768] = ebv*cd2;  ws[O_Z2+z+1536] = epv*cd2;
}

// ---------------- mem_avg output ----------------------------------------------
__global__ void k_memavg(const float* __restrict__ mp_in, const float* __restrict__ mem_prev,
                         const float* __restrict__ ws, float* __restrict__ out)
{
    const int tid = blockIdx.x*256 + threadIdx.x;  // 49152
    const int n = tid / 768;
    out[524800 + tid] = mp_in[n*9+0]*mem_prev[tid]
                      + mp_in[n*9+7]*ws[O_MEM1+tid]
                      + mp_in[n*9+8]*ws[O_MEM2+tid];
}

// ---------------- fused 9-module blend for att_stack_avg ----------------------
__global__ void __launch_bounds__(256)
k_final(const float* __restrict__ S, const float* __restrict__ P_in,
        const float* __restrict__ mp_in, const float* __restrict__ ws,
        float* __restrict__ out)
{
    const int tid = blockIdx.x*256 + threadIdx.x;  // 65536
    const int n = tid >> 10;
    const float* mp = mp_in + n*9;
    const float mp0=mp[0], mp1=mp[1], mp2=mp[2], mp3=mp[3], mp4=mp[4], mp5=mp[5], mp6=mp[6], mp78=mp[7]+mp[8];
    const float A   = ws[O_SCF+tid], B = ws[O_SCT+tid], Cs = ws[O_SCS+tid];
    const float rp  = ws[O_RP+tid],  rpb = ws[O_RPB+tid];
    const float tf1 = ws[O_T1F+tid], tf2 = ws[O_T2F+tid];
    const float c1f = ws[O_C1F+n],   c2f = ws[O_C2F+n];
    const float att2f = A*c2f + tf2;
    const float att1f = A*c1f + tf1;
    const float combf = fminf(att1f, att2f);
    const float cmn = fminf(rpb, rp), cmx = fmaxf(rpb, rp);
    float o[8];
    #pragma unroll
    for (int l = 0; l < 8; ++l) {
        const float Sv  = S[(size_t)tid*8 + l];
        const float P   = P_in[n*8+l];
        const float Pb  = ws[O_PB +n*8+l];
        const float pf  = ws[O_PF +n*8+l];
        const float pfb = ws[O_PFB+n*8+l];
        const float f1  = A*pf + Sv*(1.f-pf);               // Find stack entry
        o[l] = mp0*Sv
             + mp1*f1
             + mp2*(B*P + Sv*(1.f-P))
             + mp3*(combf*pfb + f1*(1.f-pfb))
             + mp4*(cmn*Pb + Sv*(1.f-Pb))
             + mp5*(cmx*Pb + Sv*(1.f-Pb))
             + mp6*(Cs*pf + Sv*(1.f-pf))
             + mp78*Sv;
    }
    float4* op = reinterpret_cast<float4*>(out + (size_t)tid*8);
    op[0] = make_float4(o[0], o[1], o[2], o[3]);
    op[1] = make_float4(o[4], o[5], o[6], o[7]);
}

extern "C" void kernel_launch(void* const* d_in, const int* in_sizes, int n_in,
                              void* d_out, int out_size, void* d_ws, size_t ws_size,
                              hipStream_t stream)
{
    const float* cs      = (const float*)d_in[0];
    const float* mp      = (const float*)d_in[1];
    const float* memp    = (const float*)d_in[2];
    const float* S       = (const float*)d_in[3];
    const float* P       = (const float*)d_in[4];
    const float* X       = (const float*)d_in[8];
    const float* mask    = (const float*)d_in[9];
    const float* f_ci_w  = (const float*)d_in[10]; const float* f_ci_b  = (const float*)d_in[11];
    const float* f_q_w   = (const float*)d_in[12]; const float* f_q_b   = (const float*)d_in[13];
    const float* f_k_w   = (const float*)d_in[14]; const float* f_k_b   = (const float*)d_in[15];
    const float* t_ci_w  = (const float*)d_in[16]; const float* t_ci_b  = (const float*)d_in[17];
    const float* t_q_w   = (const float*)d_in[18]; const float* t_q_b   = (const float*)d_in[19];
    const float* t_k_w   = (const float*)d_in[20]; const float* t_k_b   = (const float*)d_in[21];
    const float* sc_w    = (const float*)d_in[22]; const float* sc_b    = (const float*)d_in[23];
    const float* d1_ci_w = (const float*)d_in[24]; const float* d1_ci_b = (const float*)d_in[25];
    const float* d1_m_w  = (const float*)d_in[26]; const float* d1_m_b  = (const float*)d_in[27];
    const float* d2_ci_w = (const float*)d_in[28]; const float* d2_ci_b = (const float*)d_in[29];
    const float* d2_m_w  = (const float*)d_in[30]; const float* d2_m_b  = (const float*)d_in[31];
    float* out = (float*)d_out;
    float* ws  = (float*)d_ws;

    auto gemm = [&](const float* A, const float* B, const float* bias, float* C,
                    int M, int Nn, int K, int bT) {
        const int tasks = (M/32)*(Nn/32);
        if (bT)
            k_gemm_wmma<1><<<dim3((tasks+3)/4), dim3(128), 0, stream>>>(A, B, bias, C, M, Nn, K);
        else
            k_gemm_wmma<0><<<dim3((tasks+3)/4), dim3(128), 0, stream>>>(A, B, bias, C, M, Nn, K);
    };

    // pointer transforms + stack_ptr_avg output
    k_ptr<<<1, 64, 0, stream>>>(P, mp, ws, out);
    k_attread<<<256, 256, 0, stream>>>(S, P, ws);
    k_softmax<<<64, 256, 0, stream>>>(ws);

    // projection chains (WMMA bf16, fp32 accumulate)
    gemm(cs,          f_ci_w, f_ci_b,  ws+O_CFIND, 64, 768, 768, 0);
    gemm(ws+O_CFIND,  f_q_w,  f_q_b,   ws+O_QFIND, 64, 768, 768, 0);
    gemm(ws+O_QFIND,  f_k_w,  nullptr, ws+O_VFIND, 64, 768, 768, 1);  // v = Kw @ q
    gemm(cs,          t_ci_w, t_ci_b,  ws+O_CTR,   64, 768, 768, 0);
    gemm(ws+O_CTR,    t_q_w,  t_q_b,   ws+O_QTR,   64, 768, 768, 0);
    gemm(ws+O_QTR,    t_k_w,  nullptr, ws+O_VTR,   64, 768, 768, 1);
    gemm(cs,          d1_ci_w, d1_ci_b, ws+O_CD1,  64, 768, 768, 0);
    gemm(cs,          d2_ci_w, d2_ci_b, ws+O_CD2,  64, 768, 768, 0);
    k_qkb<<<1, 64, 0, stream>>>(ws, f_k_b, t_k_b);

    // the one memory-bound pass over image_hidden
    k_stream<<<256, 256, 0, stream>>>(X, mask, sc_w, sc_b, ws);
    k_ered<<<192, 256, 0, stream>>>(ws);

    // memory heads
    k_concat<<<192, 256, 0, stream>>>(cs, ws);
    gemm(ws+O_Z1, d1_m_w, d1_m_b, ws+O_MEM1, 64, 768, 2304, 0);
    gemm(ws+O_Z2, d2_m_w, d2_m_b, ws+O_MEM2, 64, 768, 2304, 0);
    k_memavg<<<192, 256, 0, stream>>>(mp, memp, ws, out);

    // fused module blend
    k_final<<<256, 256, 0, stream>>>(S, P, mp, ws, out);

    (void)in_sizes; (void)n_in; (void)out_size; (void)ws_size;
}
